// EEGHyperLSTM_19353122636356
// MI455X (gfx1250) — matrix operand order repaired and checked by reading, not compiled, not run on hardware
//
#include <hip/hip_runtime.h>
#include <cstdint>
#include <cstddef>

// ---------------------------------------------------------------------------
// EEG HyperLSTM for MI455X (gfx1250), wave32 + WMMA.
//
// B=128, T=512, F=25, H=256, HH=128, E=32.
// Strategy: batch rows are independent through the recurrence, so 8 workgroups
// each own 16 batch rows (M=16 == WMMA M) and run all 512 steps persistently
// with zero inter-workgroup sync. All matmuls are v_wmma_f32_16x16x32_bf16;
// weights are pre-swizzled into WMMA B-fragment layout (bf16) once per launch.
// Attention pooling is fused as an online softmax over timesteps.
// ---------------------------------------------------------------------------

typedef __attribute__((ext_vector_type(16))) __bf16 v16bf;
typedef __attribute__((ext_vector_type(8)))  float  v8f;

union BFrag {
    v16bf          v;
    unsigned short u[16];
    uint4          q[2];
};

static __device__ __forceinline__ unsigned short f2bf(float f) {
    unsigned int x = __float_as_uint(f);
    unsigned int r = (x + 0x7fffu + ((x >> 16) & 1u)) >> 16;   // RNE
    return (unsigned short)r;
}
static __device__ __forceinline__ float bf2f(unsigned short u) {
    return __uint_as_float(((unsigned int)u) << 16);
}
static __device__ __forceinline__ float sigmf(float x) {
    return 1.0f / (1.0f + __expf(-x));
}
static __device__ __forceinline__ float wave_sum(float v) {
#pragma unroll
    for (int off = 16; off > 0; off >>= 1) v += __shfl_xor(v, off, 32);
    return v;
}

// A fragment: 16(M) x 32(K) bf16 from LDS activations (row-major, ushort bf16).
// ISA layout: lane<16 holds row M=lane, K {0..7,16..23}; lane>=16 same row set,
// K {8..15,24..31}.  (k-half offset folded into the row pointer.)
static __device__ __forceinline__ v16bf load_a_frag(const unsigned short* act,
                                                    int stride, int kbase, int lane) {
    BFrag fr;
    const unsigned short* row = act + (lane & 15) * stride + kbase + ((lane >> 4) << 3);
#pragma unroll
    for (int j = 0; j < 4; ++j) {
        fr.u[2 * j]     = row[2 * j];
        fr.u[2 * j + 1] = row[2 * j + 1];
        fr.u[8 + 2 * j]     = row[16 + 2 * j];
        fr.u[8 + 2 * j + 1] = row[16 + 2 * j + 1];
    }
    return fr.v;
}

// B fragment: 32(K) x 16(N) bf16, pre-swizzled in global memory:
// element (n_local=lane&15, k_local=8*(lane>>4)+(e&7)+16*(e>>3)) at frags[(fidx*32+lane)*16+e].
// Two coalesced b128 loads per lane (32B/lane, L2 resident).
static __device__ __forceinline__ v16bf load_b_frag(const unsigned short* frags,
                                                    int fidx, int lane) {
    BFrag fr;
    const uint4* q = (const uint4*)(frags + (((long)fidx * 32 + lane) << 4));
    fr.q[0] = q[0];
    fr.q[1] = q[1];
    return fr.v;
}

static __device__ __forceinline__ v8f wmma_bf16(v16bf a, v16bf b, v8f c) {
    return __builtin_amdgcn_wmma_f32_16x16x32_bf16(false, a, false, b,
                                                   (short)0, c, false, false);
}

// ---------------------------------------------------------------------------
// Prep kernel 1: xs = relu(x @ Wp^T + bp), stored bf16 as [b*T+t][64].
// ---------------------------------------------------------------------------
__global__ __launch_bounds__(256) void xs_prep_kernel(const float* __restrict__ x,
                                                      const float* __restrict__ Wp,
                                                      const float* __restrict__ bp,
                                                      unsigned short* __restrict__ xs,
                                                      int total) {
    int idx = blockIdx.x * 256 + threadIdx.x;       // over B*T*64
    if (idx >= total) return;
    int d = idx & 63;
    long bt = idx >> 6;
    const float* xr = x + bt * 25;
    const float* w  = Wp + d * 25;
    float s = bp[d];
#pragma unroll
    for (int f = 0; f < 25; ++f) s += xr[f] * w[f];
    xs[idx] = f2bf(fmaxf(s, 0.0f));
}

// ---------------------------------------------------------------------------
// Prep kernel 2: pack a row-major (N,K) f32 weight (out_features x in_features)
// into bf16 WMMA B fragments: frag f = nt*KT + kt; element (lane,e) holds
// B[k][n] = src[n*K + k] with n = nt*16+(lane&15),
// k = kt*32 + 8*(lane>>4) + (e&7) + 16*(e>>3).
// ---------------------------------------------------------------------------
__global__ __launch_bounds__(256) void pack_b_tiles(const float* __restrict__ src,
                                                    unsigned short* __restrict__ dst,
                                                    int N, int K) {
    int KT = K >> 5;
    long total = (long)(N >> 4) * KT * 512;
    long idx = (long)blockIdx.x * 256 + threadIdx.x;
    if (idx >= total) return;
    int e    = (int)(idx & 15);
    int lane = (int)((idx >> 4) & 31);
    long f   = idx >> 9;
    int kt = (int)(f % KT);
    int nt = (int)(f / KT);
    int k = kt * 32 + ((lane >> 4) << 3) + (e & 7) + ((e >> 3) << 4);
    int n = nt * 16 + (lane & 15);
    dst[idx] = f2bf(src[(long)n * K + k]);
}

// ---------------------------------------------------------------------------
// Main persistent recurrent kernel: 8 workgroups x 256 threads (8 waves).
// Workgroup g owns batch rows [16g, 16g+16).
// ---------------------------------------------------------------------------
struct HLParams {
    const unsigned short* xs;        // [B*T][64] bf16
    const unsigned short* w_hypih;   // frags: NT=32, KT=10
    const unsigned short* w_hyphh;   // NT=32, KT=4
    const unsigned short* w_proj;    // NT=24, KT=4
    const unsigned short* w_scale;   // NT=192, KT=1
    const unsigned short* w_mainih;  // NT=64, KT=2
    const unsigned short* w_mainhh;  // NT=64, KT=8
    const unsigned short* w_att1;    // NT=8,  KT=8
    const float* hyp_ih_b;           // 512
    const float* hyp_ln_g;           // 4*128
    const float* hyp_ln_b;
    const float* hyp_lnc_g;          // 128
    const float* hyp_lnc_b;
    const float* proj_b;             // 12*32 = 384
    const float* main_b;             // 1024
    const float* ln_g;               // 4*256
    const float* ln_b;
    const float* lnc_g;              // 256
    const float* lnc_b;
    const float* att_b1;             // 128
    const float* att_W2;             // 128
    const float* att_b2;             // 1
    float*       out;                // [B][256]
};

#define SM_BYTES (16*256*2 + 16*128*2 + 16*64*2 + 16*384*2 /* bf16 region */ \
                + (16*256 + 16*128 + 16*512 + 16*1024 + 16*256 + 16*128 + 16*256 + 64) * 4)

__global__ __launch_bounds__(256, 1) void hyperlstm_kernel(HLParams p) {
    extern __shared__ char smem_raw[];
    unsigned short* sm_h  = (unsigned short*)smem_raw;      // [16][256] bf16 h
    unsigned short* sm_hh = sm_h  + 16 * 256;               // [16][128] bf16 hyper h
    unsigned short* sm_x  = sm_hh + 16 * 128;               // [16][64]  bf16 x_t
    unsigned short* sm_z  = sm_x  + 16 * 64;                // [16][384] bf16 z
    float* sm_c   = (float*)(sm_z + 16 * 384);              // [16][256] f32 c
    float* sm_hc  = sm_c   + 16 * 256;                      // [16][128] f32 hyper c
    float* sm_g4  = sm_hc  + 16 * 128;                      // [16][512] hyper preact
    float* sm_pre = sm_g4  + 16 * 512;                      // [16][1024] main preact
    float* sm_hf  = sm_pre + 16 * 1024;                     // [16][256] f32 new h
    float* sm_u   = sm_hf  + 16 * 256;                      // [16][128] attn hidden
    float* sm_acc = sm_u   + 16 * 128;                      // [16][256] softmax acc
    float* sm_m   = sm_acc + 16 * 256;                      // [16] running max
    float* sm_s   = sm_m + 16;                              // [16] running sum
    float* sm_al  = sm_s + 16;                              // [16] alpha
    float* sm_be  = sm_al + 16;                             // [16] beta

    const int tid  = threadIdx.x;
    const int lane = tid & 31;
    const int wave = tid >> 5;
    const int b0   = blockIdx.x * 16;
    const float EPS = 1e-6f;

    for (int i = tid; i < 16 * 256; i += 256) { sm_h[i] = 0; sm_c[i] = 0.0f; sm_acc[i] = 0.0f; }
    for (int i = tid; i < 16 * 128; i += 256) { sm_hh[i] = 0; sm_hc[i] = 0.0f; }
    if (tid < 16) { sm_m[tid] = -3.0e38f; sm_s[tid] = 0.0f; }
    __syncthreads();

    for (int t = 0; t < 512; ++t) {
        // ---- Phase A: stage x_t (bf16) -----------------------------------
        for (int i = tid; i < 16 * 64; i += 256) {
            int r = i >> 6, d = i & 63;
            sm_x[i] = p.xs[((long)(b0 + r) * 512 + t) * 64 + d];
        }
        __syncthreads();

        // ---- Phase B: hyper gates  g4 = [x_t,h]@Wih^T + hh@Whh^T + b -----
        for (int nt = wave; nt < 32; nt += 8) {
            v8f acc = {};
#pragma unroll
            for (int kt = 0; kt < 10; ++kt) {
                v16bf a = (kt < 2) ? load_a_frag(sm_x, 64, kt * 32, lane)
                                   : load_a_frag(sm_h, 256, (kt - 2) * 32, lane);
                acc = wmma_bf16(a, load_b_frag(p.w_hypih, nt * 10 + kt, lane), acc);
            }
#pragma unroll
            for (int kt = 0; kt < 4; ++kt) {
                v16bf a = load_a_frag(sm_hh, 128, kt * 32, lane);
                acc = wmma_bf16(a, load_b_frag(p.w_hyphh, nt * 4 + kt, lane), acc);
            }
            int n = nt * 16 + (lane & 15);
            float bias = p.hyp_ih_b[n];
            int mh = (lane >> 4) * 8;
#pragma unroll
            for (int j = 0; j < 8; ++j) sm_g4[(mh + j) * 512 + n] = acc[j] + bias;
        }
        __syncthreads();

        // ---- Phase C1: LN over g4 rows of 128 (per b,gate); hf += 1 ------
        for (int row = wave; row < 64; row += 8) {
            int b = row >> 2, g = row & 3;
            float* xr = sm_g4 + b * 512 + g * 128;
            float v0 = xr[lane], v1 = xr[lane + 32], v2 = xr[lane + 64], v3 = xr[lane + 96];
            float mean = wave_sum(v0 + v1 + v2 + v3) * (1.0f / 128.0f);
            float d0 = v0 - mean, d1 = v1 - mean, d2 = v2 - mean, d3 = v3 - mean;
            float q = wave_sum(d0 * d0 + d1 * d1 + d2 * d2 + d3 * d3);
            float inv = 1.0f / (sqrtf(q * (1.0f / 127.0f)) + EPS);
            const float* gm = p.hyp_ln_g + g * 128;
            const float* bt = p.hyp_ln_b + g * 128;
            float add = (g == 1) ? 1.0f : 0.0f;
            xr[lane]      = gm[lane]      * d0 * inv + bt[lane]      + add;
            xr[lane + 32] = gm[lane + 32] * d1 * inv + bt[lane + 32] + add;
            xr[lane + 64] = gm[lane + 64] * d2 * inv + bt[lane + 64] + add;
            xr[lane + 96] = gm[lane + 96] * d3 * inv + bt[lane + 96] + add;
        }
        __syncthreads();

        // ---- Phase C2: hyper cell + LN(c) + new hh -----------------------
        for (int b = wave; b < 16; b += 8) {
            float* gr = sm_g4 + b * 512;
            float nhc[4];
            float s = 0.0f;
#pragma unroll
            for (int u = 0; u < 4; ++u) {
                int n = lane + u * 32;
                float hi = gr[n], hf = gr[128 + n], hg = gr[256 + n];
                float v = sm_hc[b * 128 + n] * sigmf(hf) + sigmf(hi) * tanhf(hg);
                nhc[u] = v; s += v;
            }
            float mean = wave_sum(s) * (1.0f / 128.0f);
            float q = 0.0f;
#pragma unroll
            for (int u = 0; u < 4; ++u) { float d = nhc[u] - mean; q += d * d; }
            float inv = 1.0f / (sqrtf(wave_sum(q) * (1.0f / 127.0f)) + EPS);
#pragma unroll
            for (int u = 0; u < 4; ++u) {
                int n = lane + u * 32;
                float y = p.hyp_lnc_g[n] * (nhc[u] - mean) * inv + p.hyp_lnc_b[n];
                sm_hc[b * 128 + n] = y;
                float ho = gr[384 + n];
                sm_hh[b * 128 + n] = f2bf(tanhf(y) * sigmf(ho));
            }
        }
        __syncthreads();

        // ---- Phase D: z = hh @ projW + proj_b  (N=384) -------------------
        for (int nt = wave; nt < 24; nt += 8) {
            v8f acc = {};
#pragma unroll
            for (int kt = 0; kt < 4; ++kt) {
                v16bf a = load_a_frag(sm_hh, 128, kt * 32, lane);
                acc = wmma_bf16(a, load_b_frag(p.w_proj, nt * 4 + kt, lane), acc);
            }
            int n = nt * 16 + (lane & 15);
            float bias = p.proj_b[n];
            int mh = (lane >> 4) * 8;
#pragma unroll
            for (int j = 0; j < 8; ++j) sm_z[(mh + j) * 384 + n] = f2bf(acc[j] + bias);
        }
        __syncthreads();

        // ---- Phase E: fused main gates  pre = xg*v0 + hg*v1 + v2 + b -----
        for (int jt = wave; jt < 64; jt += 8) {
            int g = jt >> 4;
            v8f ax = {}, ah = {}, v0 = {}, v1 = {}, v2 = {};
#pragma unroll
            for (int kt = 0; kt < 2; ++kt)
                ax = wmma_bf16(load_a_frag(sm_x, 64, kt * 32, lane),
                               load_b_frag(p.w_mainih, jt * 2 + kt, lane), ax);
#pragma unroll
            for (int kt = 0; kt < 8; ++kt)
                ah = wmma_bf16(load_a_frag(sm_h, 256, kt * 32, lane),
                               load_b_frag(p.w_mainhh, jt * 8 + kt, lane), ah);
            v0 = wmma_bf16(load_a_frag(sm_z, 384, g * 32, lane),
                           load_b_frag(p.w_scale, jt, lane), v0);
            v1 = wmma_bf16(load_a_frag(sm_z, 384, (g + 4) * 32, lane),
                           load_b_frag(p.w_scale, jt + 64, lane), v1);
            v2 = wmma_bf16(load_a_frag(sm_z, 384, (g + 8) * 32, lane),
                           load_b_frag(p.w_scale, jt + 128, lane), v2);
            int n = jt * 16 + (lane & 15);       // n in [0,1024)
            float bias = p.main_b[n];            // f_shift is a per-row const -> LN-null
            int mh = (lane >> 4) * 8;
#pragma unroll
            for (int j = 0; j < 8; ++j)
                sm_pre[(mh + j) * 1024 + n] = ax[j] * v0[j] + ah[j] * v1[j] + v2[j] + bias;
        }
        __syncthreads();

        // ---- Phase F1: LN over pre rows of 256 ---------------------------
        for (int row = wave; row < 64; row += 8) {
            int b = row >> 2, g = row & 3;
            float* xr = sm_pre + b * 1024 + g * 256;
            float v[8];
            float s = 0.0f;
#pragma unroll
            for (int u = 0; u < 8; ++u) { v[u] = xr[lane + u * 32]; s += v[u]; }
            float mean = wave_sum(s) * (1.0f / 256.0f);
            float q = 0.0f;
#pragma unroll
            for (int u = 0; u < 8; ++u) { float d = v[u] - mean; q += d * d; }
            float inv = 1.0f / (sqrtf(wave_sum(q) * (1.0f / 255.0f)) + EPS);
            const float* gm = p.ln_g + g * 256;
            const float* bt = p.ln_b + g * 256;
#pragma unroll
            for (int u = 0; u < 8; ++u) {
                int n = lane + u * 32;
                xr[n] = gm[n] * (v[u] - mean) * inv + bt[n];
            }
        }
        __syncthreads();

        // ---- Phase F2: main cell + LN(c) + new h -------------------------
        for (int b = wave; b < 16; b += 8) {
            float* pr = sm_pre + b * 1024;
            float nc[8];
            float s = 0.0f;
#pragma unroll
            for (int u = 0; u < 8; ++u) {
                int n = lane + u * 32;
                float i_ = pr[n], f_ = pr[256 + n], g_ = pr[512 + n];
                float v = sm_c[b * 256 + n] * sigmf(f_) + sigmf(i_) * tanhf(g_);
                nc[u] = v; s += v;
            }
            float mean = wave_sum(s) * (1.0f / 256.0f);
            float q = 0.0f;
#pragma unroll
            for (int u = 0; u < 8; ++u) { float d = nc[u] - mean; q += d * d; }
            float inv = 1.0f / (sqrtf(wave_sum(q) * (1.0f / 255.0f)) + EPS);
#pragma unroll
            for (int u = 0; u < 8; ++u) {
                int n = lane + u * 32;
                float y = p.lnc_g[n] * (nc[u] - mean) * inv + p.lnc_b[n];
                sm_c[b * 256 + n] = y;
                float o_ = pr[768 + n];
                float h = tanhf(y) * sigmf(o_);
                sm_hf[b * 256 + n] = h;
                sm_h[b * 256 + n]  = f2bf(h);
            }
        }
        __syncthreads();

        // ---- Phase G1: attention hidden  u = tanh(h @ W1^T + b1) ---------
        {
            int nt = wave;                       // 8 waves == 8 n-tiles of 128
            v8f acc = {};
#pragma unroll
            for (int kt = 0; kt < 8; ++kt)
                acc = wmma_bf16(load_a_frag(sm_h, 256, kt * 32, lane),
                                load_b_frag(p.w_att1, nt * 8 + kt, lane), acc);
            int n = nt * 16 + (lane & 15);
            float b1 = p.att_b1[n];
            int mh = (lane >> 4) * 8;
#pragma unroll
            for (int j = 0; j < 8; ++j) sm_u[(mh + j) * 128 + n] = tanhf(acc[j] + b1);
        }
        __syncthreads();

        // ---- Phase G2: logit + online softmax scalars --------------------
        for (int b = wave; b < 16; b += 8) {
            float s = 0.0f;
#pragma unroll
            for (int u = 0; u < 4; ++u) {
                int n = lane + u * 32;
                s += sm_u[b * 128 + n] * p.att_W2[n];
            }
            s = wave_sum(s);
            if (lane == 0) {
                float a = s + p.att_b2[0];
                float m_old = sm_m[b];
                float m_new = fmaxf(m_old, a);
                float al = __expf(m_old - m_new);
                float be = __expf(a - m_new);
                sm_m[b]  = m_new;
                sm_s[b]  = sm_s[b] * al + be;
                sm_al[b] = al;
                sm_be[b] = be;
            }
        }
        __syncthreads();

        // ---- Phase G3: streaming weighted accumulator --------------------
        for (int i = tid; i < 16 * 256; i += 256) {
            int b = i >> 8;
            sm_acc[i] = sm_acc[i] * sm_al[b] + sm_hf[i] * sm_be[b];
        }
        __syncthreads();
    }

    // Final: out = acc / s
    for (int i = tid; i < 16 * 256; i += 256) {
        int b = i >> 8, n = i & 255;
        p.out[(long)(b0 + b) * 256 + n] = sm_acc[i] / sm_s[b];
    }
}

// ---------------------------------------------------------------------------
// Host launcher
// ---------------------------------------------------------------------------
extern "C" void kernel_launch(void* const* d_in, const int* in_sizes, int n_in,
                              void* d_out, int out_size, void* d_ws, size_t ws_size,
                              hipStream_t stream) {
    (void)in_sizes; (void)n_in; (void)out_size; (void)ws_size;
    const int B = 128, T = 512;

    const float* x         = (const float*)d_in[0];
    const float* Wp        = (const float*)d_in[1];
    const float* bp        = (const float*)d_in[2];
    const float* hyp_ih_W  = (const float*)d_in[3];
    const float* hyp_ih_b  = (const float*)d_in[4];
    const float* hyp_hh_W  = (const float*)d_in[5];
    const float* hyp_ln_g  = (const float*)d_in[6];
    const float* hyp_ln_b  = (const float*)d_in[7];
    const float* hyp_lnc_g = (const float*)d_in[8];
    const float* hyp_lnc_b = (const float*)d_in[9];
    const float* proj_W    = (const float*)d_in[10];
    const float* proj_b    = (const float*)d_in[11];
    const float* scale_W   = (const float*)d_in[12];
    const float* main_ih_W = (const float*)d_in[13];
    const float* main_hh_W = (const float*)d_in[14];
    const float* main_b    = (const float*)d_in[15];
    const float* ln_g      = (const float*)d_in[16];
    const float* ln_b      = (const float*)d_in[17];
    const float* lnc_g     = (const float*)d_in[18];
    const float* lnc_b     = (const float*)d_in[19];
    const float* att_W1    = (const float*)d_in[20];
    const float* att_b1    = (const float*)d_in[21];
    const float* att_W2    = (const float*)d_in[22];
    const float* att_b2    = (const float*)d_in[23];

    // ---- workspace carve (all bf16/ushort regions) ----
    char* ws = (char*)d_ws;
    size_t off = 0;
    auto take = [&](size_t bytes) -> char* {
        off = (off + 255) & ~(size_t)255;
        char* pp = ws + off;
        off += bytes;
        return pp;
    };
    unsigned short* xs_bf     = (unsigned short*)take((size_t)B * T * 64 * 2);
    unsigned short* w_hypih   = (unsigned short*)take((size_t)(512/16) * (320/32) * 512 * 2);
    unsigned short* w_hyphh   = (unsigned short*)take((size_t)(512/16) * (128/32) * 512 * 2);
    unsigned short* w_proj    = (unsigned short*)take((size_t)(384/16) * (128/32) * 512 * 2);
    unsigned short* w_scale   = (unsigned short*)take((size_t)(3072/16) * (32/32) * 512 * 2);
    unsigned short* w_mainih  = (unsigned short*)take((size_t)(1024/16) * (64/32) * 512 * 2);
    unsigned short* w_mainhh  = (unsigned short*)take((size_t)(1024/16) * (256/32) * 512 * 2);
    unsigned short* w_att1    = (unsigned short*)take((size_t)(128/16) * (256/32) * 512 * 2);

    // ---- prep: xs projection ----
    {
        int total = B * T * 64;
        xs_prep_kernel<<<(total + 255) / 256, 256, 0, stream>>>(x, Wp, bp, xs_bf, total);
    }
    // ---- prep: weight swizzle into WMMA B fragments ----
    auto pack = [&](const float* src, unsigned short* dst, int N, int K) {
        long total = (long)(N / 16) * (K / 32) * 512;
        pack_b_tiles<<<(unsigned)((total + 255) / 256), 256, 0, stream>>>(src, dst, N, K);
    };
    pack(hyp_ih_W,  w_hypih,  512,  320);
    pack(hyp_hh_W,  w_hyphh,  512,  128);
    pack(proj_W,    w_proj,   384,  128);   // (12,32,128) == (384,128) row-major
    pack(scale_W,   w_scale,  3072, 32);    // (12,256,32) == (3072,32) row-major
    pack(main_ih_W, w_mainih, 1024, 64);
    pack(main_hh_W, w_mainhh, 1024, 256);
    pack(att_W1,    w_att1,   128,  256);

    // ---- main persistent kernel ----
    HLParams p;
    p.xs = xs_bf;
    p.w_hypih = w_hypih;   p.w_hyphh = w_hyphh; p.w_proj = w_proj;
    p.w_scale = w_scale;   p.w_mainih = w_mainih; p.w_mainhh = w_mainhh;
    p.w_att1 = w_att1;
    p.hyp_ih_b = hyp_ih_b;
    p.hyp_ln_g = hyp_ln_g;   p.hyp_ln_b = hyp_ln_b;
    p.hyp_lnc_g = hyp_lnc_g; p.hyp_lnc_b = hyp_lnc_b;
    p.proj_b = proj_b;       p.main_b = main_b;
    p.ln_g = ln_g;           p.ln_b = ln_b;
    p.lnc_g = lnc_g;         p.lnc_b = lnc_b;
    p.att_b1 = att_b1;       p.att_W2 = att_W2;   p.att_b2 = att_b2;
    p.out = (float*)d_out;

    // ~186 KB dynamic LDS: legal on CDNA5 (320 KB/WGP) but above default cap.
    hipFuncSetAttribute((const void*)hyperlstm_kernel,
                        hipFuncAttributeMaxDynamicSharedMemorySize, SM_BYTES);
    hyperlstm_kernel<<<B / 16, 256, SM_BYTES, stream>>>(p);
}
